// FeatureFusion_84645215470251
// MI455X (gfx1250) — matrix operand order repaired and checked by reading, not compiled, not run on hardware
//
#include <hip/hip_runtime.h>

typedef __attribute__((ext_vector_type(16))) __bf16 v16bf;
typedef __attribute__((ext_vector_type(8)))  float  v8f;

#if defined(__has_builtin)
#  if __has_builtin(__builtin_amdgcn_global_load_async_to_lds_b128)
#    define USE_ASYNC_LDS 1
#  endif
#endif
#ifndef USE_ASYNC_LDS
#  define USE_ASYNC_LDS 0
#endif

union Frag32 {
    v16bf v;
    uint4 q[2];
    unsigned short s[16];
};

static __device__ __forceinline__ unsigned short f2bf(float f) {
    union { float f; unsigned int u; } v;
    v.f = f;
    unsigned int r = v.u + 0x7FFFu + ((v.u >> 16) & 1u);  // round-to-nearest-even
    return (unsigned short)(r >> 16);
}

static __device__ __forceinline__ void wait_async_copies() {
#if USE_ASYNC_LDS
#  if __has_builtin(__builtin_amdgcn_s_wait_asynccnt)
    __builtin_amdgcn_s_wait_asynccnt(0);
#  else
    asm volatile("s_wait_asynccnt 0" ::: "memory");
#  endif
#endif
}

// ---------------------------------------------------------------------------
// Prep kernels
// ---------------------------------------------------------------------------

__global__ void inv_init_kernel(int* __restrict__ inv, long total, int n) {
    long i = (long)blockIdx.x * blockDim.x + threadIdx.x;
    if (i < total) inv[i] = n;   // sentinel -> zero pad row
}

__global__ void inv_scatter_kernel(const int* __restrict__ in_idx,
                                   const int* __restrict__ out_idx,
                                   int* __restrict__ inv, long total, int n) {
    long i = (long)blockIdx.x * blockDim.x + threadIdx.x;
    if (i >= total) return;
    int o = out_idx[i];
    if (o < n) {
        long k = i / n;
        inv[k * n + o] = in_idx[i];
    }
}

// xg -> buf0 (bf16), x_F -> cat[:, :128] (bf16)
__global__ void convert_feat_kernel(const float* __restrict__ xF,
                                    const float* __restrict__ xg,
                                    unsigned short* __restrict__ buf0,
                                    unsigned short* __restrict__ cat,
                                    long total) {
    long i = (long)blockIdx.x * blockDim.x + threadIdx.x;
    if (i >= total) return;
    long row = i >> 7;
    int  c   = (int)(i & 127);
    buf0[i] = f2bf(xg[i]);
    cat[row * 256 + c] = f2bf(xF[i]);
}

// zero sentinel pad rows (row N of buf0, buf1, cat)
__global__ void zero_pad_rows_kernel(unsigned short* __restrict__ buf0,
                                     unsigned short* __restrict__ buf1,
                                     unsigned short* __restrict__ cat, int n) {
    int t = threadIdx.x;   // 256 threads
    if (t < 128) {
        buf0[(size_t)n * 128 + t] = 0;
        buf1[(size_t)n * 128 + t] = 0;
    }
    cat[(size_t)n * 256 + t] = 0;
}

// Pack W[k][c][nout] (f32, row-major) into per-lane WMMA B fragments (bf16):
//   Wp[(((k*KT + kt)*8 + nt)*32 + lane)*16 + e] = W[k][kt*32 + (lane>>4)*16 + e][nt*16 + (lane&15)]
__global__ void pack_weights_kernel(const float* __restrict__ W,
                                    unsigned short* __restrict__ Wp,
                                    int cin, long total) {
    long i = (long)blockIdx.x * blockDim.x + threadIdx.x;
    if (i >= total) return;
    int KT = cin >> 5;
    int e = (int)(i & 15);
    long r = i >> 4;
    int lane = (int)(r & 31); r >>= 5;
    int nt   = (int)(r & 7);  r >>= 3;
    int kt   = (int)(r % KT);
    int k    = (int)(r / KT);
    int c    = kt * 32 + (lane >> 4) * 16 + e;
    int ncol = nt * 16 + (lane & 15);
    Wp[i] = f2bf(W[((size_t)k * cin + c) * 128 + ncol]);
}

// ---------------------------------------------------------------------------
// Fused gather + WMMA GEMM + bias(+relu) conv pass.
// 32 output rows per workgroup; 8 waves = 8 N-tiles; 2 accumulators per wave
// (rows 0-15 / 16-31) so every packed-B fragment feeds two independent WMMAs.
// Staging of the next offset's gathered rows overlaps compute via
// GLOBAL_LOAD_ASYNC_TO_LDS_B128 (ASYNCcnt) when the builtin is available.
//   MODE 0: bias, store bf16 to outBf (stride 128)            [conv0 -> buf1]
//   MODE 1: bias, relu, store bf16 to outBf+128 (stride 256)  [conv1 -> cat]
//   MODE 2: bias, store f32 to outF (stride 128)              [conv2 -> d_out]
// ---------------------------------------------------------------------------
template <int CIN, int MODE>
__global__ __launch_bounds__(256) void sparse_conv_wmma_kernel(
        const unsigned short* __restrict__ bufIn,  // (n+1) x CIN, bf16
        const unsigned short* __restrict__ Wp,     // packed weights
        const float* __restrict__ bias,            // 128
        const int* __restrict__ inv,               // 27 x n
        unsigned short* __restrict__ outBf,
        float* __restrict__ outF,
        int n) {
    constexpr int KT = CIN / 32;
    __shared__ unsigned short smem[2][32 * CIN];

    const int t    = threadIdx.x;
    const int lane = t & 31;
    const int wave = t >> 5;          // = output N-tile (0..7)
    const int row0 = blockIdx.x * 32;
    const int h    = lane >> 4;

    // 8 threads per gathered row; each copies one contiguous 32/64B segment.
    auto stage = [&](int buf, int k) {
        int r  = t >> 3;              // gathered row 0..31
        int rg = row0 + r;
        int idx = (rg < n) ? inv[(size_t)k * n + rg] : n;
        const uint4* src = (const uint4*)(bufIn + (size_t)idx * CIN) + (t & 7) * (CIN / 64);
        uint4*       dst = (uint4*)(smem[buf] + r * CIN)            + (t & 7) * (CIN / 64);
#if USE_ASYNC_LDS
        typedef int v4i_t __attribute__((vector_size(16)));
        typedef __attribute__((address_space(1))) v4i_t* gv4i_p;
        typedef __attribute__((address_space(3))) v4i_t* lv4i_p;
#pragma unroll
        for (int c = 0; c < CIN / 64; ++c) {
            // generic->AS casts via integer round-trip: AS(3) ptrs are the low
            // 32 bits of the generic LDS address (aperture lives in [63:32]).
            __builtin_amdgcn_global_load_async_to_lds_b128(
                (gv4i_p)(unsigned long long)(src + c),
                (lv4i_p)(unsigned long long)(dst + c), 0, 0);
        }
#else
#pragma unroll
        for (int c = 0; c < CIN / 64; ++c) dst[c] = src[c];
#endif
    };

    v8f acc0 = {};
    v8f acc1 = {};
    stage(0, 0);
    wait_async_copies();
    __syncthreads();

    for (int k = 0; k < 27; ++k) {
        if (k + 1 < 27) stage((k + 1) & 1, k + 1);

        const unsigned short* base  = smem[k & 1];
        const unsigned short* a0row = base + (lane & 15) * CIN;
        const unsigned short* a1row = base + (16 + (lane & 15)) * CIN;
#pragma unroll
        for (int kt = 0; kt < KT; ++kt) {
            Frag32 a0, a1, b;
            // B: one contiguous 32-byte pre-packed fragment per lane
            const uint4* bp = (const uint4*)(Wp +
                ((((size_t)k * KT + kt) * 8 + wave) * 32 + lane) * 16);
            b.q[0] = bp[0];
            b.q[1] = bp[1];
            // A: row m = lane&15 (+16); K = {16h..16h+15} in this K-tile
            const uint4* ap0 = (const uint4*)(a0row + kt * 32 + 16 * h);
            a0.q[0] = ap0[0];
            a0.q[1] = ap0[1];
            const uint4* ap1 = (const uint4*)(a1row + kt * 32 + 16 * h);
            a1.q[0] = ap1[0];
            a1.q[1] = ap1[1];
            acc0 = __builtin_amdgcn_wmma_f32_16x16x32_bf16(
                false, a0.v, false, b.v, (short)0, acc0, false, false);
            acc1 = __builtin_amdgcn_wmma_f32_16x16x32_bf16(
                false, a1.v, false, b.v, (short)0, acc1, false, false);
        }
        wait_async_copies();   // next-buffer async writes landed in LDS
        __syncthreads();
    }

    // Epilogue: lane's column is fixed (col = wave*16 + lane&15); rows m = r + 8h.
    const int col = wave * 16 + (lane & 15);
    const float bv = bias[col];
#pragma unroll
    for (int sub = 0; sub < 2; ++sub) {
        const v8f& acc = sub ? acc1 : acc0;
#pragma unroll
        for (int r = 0; r < 8; ++r) {
            int m = row0 + sub * 16 + r + 8 * h;
            if (m >= n) continue;
            float v = acc[r] + bv;
            if (MODE == 1) v = v > 0.0f ? v : 0.0f;
            if (MODE == 2) {
                outF[(size_t)m * 128 + col] = v;
            } else if (MODE == 1) {
                outBf[(size_t)m * 256 + 128 + col] = f2bf(v);
            } else {
                outBf[(size_t)m * 128 + col] = f2bf(v);
            }
        }
    }
}

// ---------------------------------------------------------------------------

extern "C" void kernel_launch(void* const* d_in, const int* in_sizes, int n_in,
                              void* d_out, int out_size, void* d_ws, size_t ws_size,
                              hipStream_t stream) {
    const float* xF = (const float*)d_in[0];
    const float* xg = (const float*)d_in[1];
    const float* W0 = (const float*)d_in[2];
    const float* b0 = (const float*)d_in[3];
    const float* W1 = (const float*)d_in[4];
    const float* b1 = (const float*)d_in[5];
    const float* W2 = (const float*)d_in[6];
    const float* b2 = (const float*)d_in[7];
    const int* in_idx  = (const int*)d_in[8];
    const int* out_idx = (const int*)d_in[9];

    const int n = in_sizes[0] / 128;       // N voxels
    const long mapTotal = 27L * n;

    // workspace carve-out (256B-aligned slices)
    char* ws = (char*)d_ws;
    size_t off = 0;
    auto alloc = [&](size_t bytes) {
        char* p = ws + off;
        off = (off + bytes + 255) & ~(size_t)255;
        return p;
    };
    int*            inv  = (int*)           alloc(sizeof(int) * (size_t)27 * n);
    unsigned short* buf0 = (unsigned short*)alloc(2ull * (n + 1) * 128);
    unsigned short* buf1 = (unsigned short*)alloc(2ull * (n + 1) * 128);
    unsigned short* cat  = (unsigned short*)alloc(2ull * (n + 1) * 256);
    unsigned short* Wp0  = (unsigned short*)alloc(2ull * 27 * 128 * 128);
    unsigned short* Wp1  = (unsigned short*)alloc(2ull * 27 * 128 * 128);
    unsigned short* Wp2  = (unsigned short*)alloc(2ull * 27 * 256 * 128);
    (void)ws_size;

    const int B = 256;
    // 1) invert kernel maps (scatter -> gather form)
    inv_init_kernel<<<(int)((mapTotal + B - 1) / B), B, 0, stream>>>(inv, mapTotal, n);
    inv_scatter_kernel<<<(int)((mapTotal + B - 1) / B), B, 0, stream>>>(in_idx, out_idx, inv, mapTotal, n);

    // 2) features -> bf16 (xg -> buf0, x_F -> cat[:, :128]); zero pad rows
    long feTotal = (long)n * 128;
    convert_feat_kernel<<<(int)((feTotal + B - 1) / B), B, 0, stream>>>(xF, xg, buf0, cat, feTotal);
    zero_pad_rows_kernel<<<1, 256, 0, stream>>>(buf0, buf1, cat, n);

    // 3) pack weights into WMMA B-fragment layout
    long w01Total = 27L * 128 * 128;
    long w2Total  = 27L * 256 * 128;
    pack_weights_kernel<<<(int)((w01Total + B - 1) / B), B, 0, stream>>>(W0, Wp0, 128, w01Total);
    pack_weights_kernel<<<(int)((w01Total + B - 1) / B), B, 0, stream>>>(W1, Wp1, 128, w01Total);
    pack_weights_kernel<<<(int)((w2Total  + B - 1) / B), B, 0, stream>>>(W2, Wp2, 256, w2Total);

    // 4) three conv passes, 32 output rows per workgroup, 8 waves = 8 N-tiles
    int tiles = (n + 31) / 32;
    sparse_conv_wmma_kernel<128, 0><<<tiles, 256, 0, stream>>>(buf0, Wp0, b0, inv, buf1, nullptr, n);
    sparse_conv_wmma_kernel<128, 1><<<tiles, 256, 0, stream>>>(buf1, Wp1, b1, inv, cat,  nullptr, n);
    sparse_conv_wmma_kernel<256, 2><<<tiles, 256, 0, stream>>>(cat,  Wp2, b2, inv, nullptr, (float*)d_out, n);
}
